// self_attention_88716844466519
// MI455X (gfx1250) — matrix operand order repaired and checked by reading, not compiled, not run on hardware
//
#include <hip/hip_runtime.h>
#include <hip/hip_bf16.h>
#include <stdint.h>

typedef __bf16 bf16_t;
typedef __attribute__((ext_vector_type(16))) __bf16 v16bf;
typedef __attribute__((ext_vector_type(8)))  float  v8f;
typedef unsigned int u32x4 __attribute__((ext_vector_type(4)));
typedef int          i32x4 __attribute__((ext_vector_type(4)));
typedef int          i32x8 __attribute__((ext_vector_type(8)));

#define CH   256
#define FG   32
#define NTOK 4096
#define NB   4
#define YROW 64            // y row = f(32) | g(32) bf16
#define OTOT 320           // f|g|h projection outputs
#define NJB  (NTOK / 32)   // key blocks

#if defined(__AMDGCN__) && __has_builtin(__builtin_amdgcn_tensor_load_to_lds)
#define USE_TDM 1
#else
#define USE_TDM 0
#endif

#if USE_TDM
// Build a 2-D Tensor-DMA descriptor (D#) per CDNA5 ISA ch.8 and issue the load.
// dims/strides in data_size(=2B) units; lds_off = raw LDS byte offset.
__device__ __forceinline__ void tdm_load_2d(uint32_t lds_off, const void* gaddr,
                                            uint32_t tensor_d0, uint32_t tensor_d1,
                                            uint32_t tile_d0, uint32_t tile_d1,
                                            uint32_t stride0) {
  uint64_t ga = (uint64_t)(uintptr_t)gaddr;
  u32x4 g0;
  g0.x = 1u;                                    // count=1 (valid user descriptor)
  g0.y = lds_off;                               // lds_addr (bytes)
  g0.z = (uint32_t)ga;                          // global_addr[31:0]
  g0.w = (uint32_t)(ga >> 32) | (2u << 30);     // global_addr[56:32] | type=2
  i32x8 g1;
  g1[0] = (int)(1u << 16);                      // wg_mask=0, data_size=1 (2 bytes)
  g1[1] = (int)((tensor_d0 & 0xffffu) << 16);   // [63:48]=tensor_dim0 lo16
  g1[2] = (int)((tensor_d0 >> 16) | ((tensor_d1 & 0xffffu) << 16));
  g1[3] = (int)((tensor_d1 >> 16) | (tile_d0 << 16));   // [127:112]=tile_dim0
  g1[4] = (int)tile_d1;                         // [143:128]=tile_dim1, tile_dim2=0
  g1[5] = (int)stride0;                         // tensor_dim0_stride lo32
  g1[6] = 0;
  g1[7] = 0;
  i32x4 gz = (i32x4){0, 0, 0, 0};
#if __clang_major__ >= 23
  i32x8 gz8 = (i32x8){0, 0, 0, 0, 0, 0, 0, 0};
  __builtin_amdgcn_tensor_load_to_lds(g0, g1, gz, gz, gz8, 0);
#else
  __builtin_amdgcn_tensor_load_to_lds(g0, g1, gz, gz, 0);
#endif
}
#endif

// ---------------------------------------------------------------------------
// Kernel 1: pack Wf|Wg|Wh into one bf16 weight block (320x256) + bias vec.
// ---------------------------------------------------------------------------
__global__ void prep_weights(const float* __restrict__ Wf, const float* __restrict__ bf_,
                             const float* __restrict__ Wg, const float* __restrict__ bg_,
                             const float* __restrict__ Wh, const float* __restrict__ bh_,
                             bf16_t* __restrict__ Wcat, float* __restrict__ bcat) {
  int tid = blockIdx.x * blockDim.x + threadIdx.x;
  int total = OTOT * CH;
  for (int i = tid; i < total; i += gridDim.x * blockDim.x) {
    int o = i / CH, c = i % CH;
    float v;
    if (o < FG)          v = Wf[o * CH + c];
    else if (o < 2 * FG) v = Wg[(o - FG) * CH + c];
    else                 v = Wh[(o - 2 * FG) * CH + c];
    Wcat[i] = (bf16_t)v;
  }
  if (tid < OTOT) {
    float v;
    if (tid < FG)          v = bf_[tid];
    else if (tid < 2 * FG) v = bg_[tid - FG];
    else                   v = bh_[tid - 2 * FG];
    bcat[tid] = v;
  }
}

// ---------------------------------------------------------------------------
// Kernel 2: fused projections via WMMA.
// 8 waves/block; wave w owns one 16-token tile. The 320x32 Wcat K-slice is
// TDM-staged into LDS once per K-step (double-buffered) and shared by all
// waves; each wave keeps a private x-transpose tile (same-wave DS ordering).
//   y[b][n][0:64] = [f|g](n)   (token-major)   hJ[b][c][n] = h(n) (channel-major)
// ---------------------------------------------------------------------------
__global__ __launch_bounds__(256) void proj_kernel(const float* __restrict__ x,
                                                   const bf16_t* __restrict__ Wcat,
                                                   const float* __restrict__ bcat,
                                                   bf16_t* __restrict__ yT,
                                                   bf16_t* __restrict__ hJ) {
  __shared__ __align__(16) bf16_t Wt[2][OTOT * 32];  // [o][c] K-slice, dbl-buffered
  __shared__ __align__(16) bf16_t xS[8][16 * 32];    // per-wave x^T tile / transposer
  const int tid  = threadIdx.x;
  const int w    = tid >> 5;
  const int lane = tid & 31;
  const int l15  = lane & 15;
  const int half = lane >> 4;
  const int blk  = blockIdx.x;
  const int b    = blk >> 5;                  // 32 blocks per batch (128 tokens)
  const int n0   = (blk & 31) * 128 + w * 16;

#if USE_TDM
  const uint32_t wt_off0 = (uint32_t)(uintptr_t)(void*)&Wt[0][0];
  const uint32_t wt_off1 = (uint32_t)(uintptr_t)(void*)&Wt[1][0];
  if (w == 0)   // prologue: slice kc=0
    tdm_load_2d(wt_off0, Wcat, CH, OTOT, 32, OTOT, CH);
#endif

  v8f acc[20];
#pragma unroll
  for (int ot = 0; ot < 20; ++ot) {
    float bl = bcat[ot * 16 + l15];
    acc[ot] = (v8f){bl, bl, bl, bl, bl, bl, bl, bl};
  }

  const size_t xbase = (size_t)b * CH * NTOK;
  for (int kc = 0; kc < 8; ++kc) {            // K loop over 256 channels, step 32
    const int c0 = kc * 32;
    const int buf = kc & 1;
#if USE_TDM
    if (w == 0) {
      if (kc + 1 < 8) {                       // prefetch next slice, dbl-buffered
        tdm_load_2d(buf ? wt_off0 : wt_off1,
                    (const char*)Wcat + (size_t)(kc + 1) * 32 * 2, CH, OTOT, 32, OTOT, CH);
        __builtin_amdgcn_s_wait_tensorcnt(1); // oldest (current slice) complete
      } else {
        __builtin_amdgcn_s_wait_tensorcnt(0);
      }
    }
#else
    for (int r = tid; r < OTOT; r += 256) {   // manual slice copy: 64 B rows
      const char* src = (const char*)Wcat + ((size_t)r * CH + c0) * 2;
      char* dst = (char*)&Wt[buf][0] + r * 64;
#pragma unroll
      for (int q = 0; q < 4; ++q)
        *(uint4*)(dst + q * 16) = *(const uint4*)(src + q * 16);
    }
#endif
    // private x tile: 32c x 16n fp32 -> bf16 LDS transpose (no barrier needed)
#pragma unroll
    for (int i = 0; i < 16; ++i) {
      int idx = i * 32 + lane;
      int cc = idx >> 4, nn = idx & 15;
      float v = x[xbase + (size_t)(c0 + cc) * NTOK + (n0 + nn)];
      xS[w][nn * 32 + cc] = (bf16_t)v;
    }
    union { v16bf v; uint4 q[2]; } fa;        // A frag: row n = l15
    {
      const char* base = (const char*)&xS[w][0] + l15 * 64 + (half ? 16 : 0);
      fa.q[0] = *(const uint4*)(base);
      fa.q[1] = *(const uint4*)(base + 32);
    }
    __syncthreads();                          // Wt[buf] ready for everyone
#pragma unroll
    for (int ot = 0; ot < 20; ++ot) {
      union { v16bf v; uint4 q[2]; } fb;      // B frag: col o = 16*ot + l15, K = c
      const char* wb = (const char*)&Wt[buf][0] + (ot * 16 + l15) * 64 + (half ? 16 : 0);
      fb.q[0] = *(const uint4*)(wb);
      fb.q[1] = *(const uint4*)(wb + 32);
      acc[ot] = __builtin_amdgcn_wmma_f32_16x16x32_bf16(false, fa.v, false, fb.v,
                                                        (short)0, acc[ot], false, false);
    }
    __syncthreads();                          // done with Wt[buf] before reuse
  }

  // f|g tiles: token-major store (coalesced along o)
#pragma unroll
  for (int ot = 0; ot < 4; ++ot) {
#pragma unroll
    for (int r = 0; r < 8; ++r) {
      int n = n0 + r + 8 * half;
      yT[((size_t)b * NTOK + n) * YROW + ot * 16 + l15] = (bf16_t)acc[ot][r];
    }
  }
  // h tiles: transpose through private LDS so hJ stores are 16B coalesced
#pragma unroll
  for (int ot = 4; ot < 20; ++ot) {
#pragma unroll
    for (int r = 0; r < 8; ++r)
      xS[w][l15 * 16 + r + 8 * half] = (bf16_t)acc[ot][r];
    int cl = lane >> 1, chunk = lane & 1;
    uint4 q = *(const uint4*)((const char*)&xS[w][0] + cl * 32 + chunk * 16);
    int c = (ot - 4) * 16 + cl;
    *(uint4*)((char*)hJ + ((size_t)(b * CH + c) * NTOK + n0 + chunk * 8) * 2) = q;
  }
}

// ---------------------------------------------------------------------------
// Kernel 3: flash attention + gamma residual epilogue.
// Block = 8 waves; wave w owns query rows [i0, i0+16) x all 256 channels.
// Key/value tiles double-buffered through LDS by the Tensor Data Mover:
// issue block jb+1, s_wait_tensorcnt(2) retires block jb (TDM is in-order).
// ---------------------------------------------------------------------------
__global__ __launch_bounds__(256) void attn_kernel(const float* __restrict__ x,
                                                   const bf16_t* __restrict__ yT,
                                                   const bf16_t* __restrict__ hJ,
                                                   const float* __restrict__ gammap,
                                                   float* __restrict__ out) {
  __shared__ __align__(16) bf16_t fS[2][32 * 32];     // f tile  [j][c]
  __shared__ __align__(16) bf16_t hS[2][CH * 32];     // h tile  [c][j]
  __shared__ __align__(16) bf16_t pS[8][16 * 32];     // per-wave P tile [i][j]
  const int tid  = threadIdx.x;
  const int w    = tid >> 5;
  const int lane = tid & 31;
  const int l15  = lane & 15;
  const int half = lane >> 4;
  const int b    = blockIdx.x >> 5;                   // 32 row-blocks per batch
  const int i0   = (blockIdx.x & 31) * 128 + w * 16;

  const bf16_t* hRow = hJ + (size_t)b * CH * NTOK;
  const bf16_t* yRow = yT + (size_t)b * NTOK * YROW;

#if USE_TDM
  const uint32_t hs_off[2] = { (uint32_t)(uintptr_t)(void*)&hS[0][0],
                               (uint32_t)(uintptr_t)(void*)&hS[1][0] };
  const uint32_t fs_off[2] = { (uint32_t)(uintptr_t)(void*)&fS[0][0],
                               (uint32_t)(uintptr_t)(void*)&fS[1][0] };
  if (w == 0) {   // prologue: stage block jb=0 into buffer 0
    tdm_load_2d(hs_off[0], hRow, NTOK, CH, 32, CH, NTOK);
    tdm_load_2d(fs_off[0], yRow, YROW, NTOK, 32, 32, YROW);
  }
#endif

  // g^T A-fragment for this wave's 16 query rows (constant over the j loop).
  union { v16bf v; uint4 q[2]; } ga;
  {
    const char* base = (const char*)(yRow + (size_t)(i0 + l15) * YROW)
                     + FG * 2 + (half ? 16 : 0);
    ga.q[0] = *(const uint4*)(base);
    ga.q[1] = *(const uint4*)(base + 32);
  }

  float m[8], l[8];
  v8f acc[16];
#pragma unroll
  for (int r = 0; r < 8; ++r) { m[r] = -1e30f; l[r] = 0.0f; }
#pragma unroll
  for (int ct = 0; ct < 16; ++ct) acc[ct] = (v8f){0, 0, 0, 0, 0, 0, 0, 0};

  for (int jb = 0; jb < NJB; ++jb) {
    const int buf = jb & 1;
#if USE_TDM
    if (w == 0) {
      if (jb + 1 < NJB) {                     // prefetch next block, dbl-buffered
        const int j1 = (jb + 1) * 32;
        tdm_load_2d(hs_off[buf ^ 1], hRow + j1, NTOK, CH, 32, CH, NTOK);
        tdm_load_2d(fs_off[buf ^ 1], yRow + (size_t)j1 * YROW, YROW, NTOK, 32, 32, YROW);
        __builtin_amdgcn_s_wait_tensorcnt(2); // two oldest = current block done
      } else {
        __builtin_amdgcn_s_wait_tensorcnt(0);
      }
    }
#else
    {   // f tile: 8 B per thread straight copy
      const int j0 = jb * 32;
      int e0 = tid * 4;
      int j = e0 >> 5, c = e0 & 31;
      const char* src = (const char*)(yRow + (size_t)(j0 + j) * YROW) + c * 2;
      *(uint2*)((char*)&fS[buf][0] + (j * 32 + c) * 2) = *(const uint2*)src;
      // h tile: thread t copies 64 B row of channel t (channel-major layout)
      const char* hsrc = (const char*)(hRow + (size_t)tid * NTOK + j0);
#pragma unroll
      for (int q = 0; q < 4; ++q)
        *(uint4*)((char*)&hS[buf][0] + tid * 64 + q * 16) = *(const uint4*)(hsrc + q * 16);
    }
#endif
    __syncthreads();                          // staged tiles visible to all waves

    // ---- scores: s = g^T f (K = 32 channels per WMMA) --------------------
    union { v16bf v; uint4 q[2]; } fb0, fb1;
    {
      const char* b0 = (const char*)&fS[buf][0] + (l15) * 64 + (half ? 16 : 0);
      fb0.q[0] = *(const uint4*)(b0);  fb0.q[1] = *(const uint4*)(b0 + 32);
      const char* b1 = (const char*)&fS[buf][0] + (16 + l15) * 64 + (half ? 16 : 0);
      fb1.q[0] = *(const uint4*)(b1);  fb1.q[1] = *(const uint4*)(b1 + 32);
    }
    const v8f zero = (v8f){0, 0, 0, 0, 0, 0, 0, 0};
    v8f s0 = __builtin_amdgcn_wmma_f32_16x16x32_bf16(false, ga.v, false, fb0.v, (short)0, zero, false, false);
    v8f s1 = __builtin_amdgcn_wmma_f32_16x16x32_bf16(false, ga.v, false, fb1.v, (short)0, zero, false, false);

    // ---- online softmax (each row lives across a 16-lane half) -----------
    float scale[8];
#pragma unroll
    for (int r = 0; r < 8; ++r) {
      float t = fmaxf(s0[r], s1[r]);
      t = fmaxf(t, __shfl_xor(t, 1));
      t = fmaxf(t, __shfl_xor(t, 2));
      t = fmaxf(t, __shfl_xor(t, 4));
      t = fmaxf(t, __shfl_xor(t, 8));
      float mn = fmaxf(m[r], t);
      scale[r] = __expf(m[r] - mn);
      m[r] = mn;
      s0[r] = __expf(s0[r] - mn);
      s1[r] = __expf(s1[r] - mn);
      float su = s0[r] + s1[r];
      su += __shfl_xor(su, 1);
      su += __shfl_xor(su, 2);
      su += __shfl_xor(su, 4);
      su += __shfl_xor(su, 8);
      l[r] = l[r] * scale[r] + su;
    }
#pragma unroll
    for (int ct = 0; ct < 16; ++ct)
#pragma unroll
      for (int r = 0; r < 8; ++r)
        acc[ct][r] *= scale[r];

    // ---- P tile: C-layout -> per-wave LDS -> A-layout --------------------
    bf16_t* pw = &pS[w][0];
#pragma unroll
    for (int r = 0; r < 8; ++r) {
      int row = r + 8 * half;
      pw[row * 32 + l15]      = (bf16_t)s0[r];
      pw[row * 32 + 16 + l15] = (bf16_t)s1[r];
    }
    union { v16bf v; uint4 q[2]; } pa;
    {
      const char* base = (const char*)pw + l15 * 64 + (half ? 16 : 0);
      pa.q[0] = *(const uint4*)(base);
      pa.q[1] = *(const uint4*)(base + 32);
    }

    // ---- o += P @ h^T : 16 WMMAs (all 256 channels) ----------------------
#pragma unroll
    for (int ct = 0; ct < 16; ++ct) {
      union { v16bf v; uint4 q[2]; } hb;   // B frag: col c = 16*ct + l15, K = j
      const char* base = (const char*)&hS[buf][0] + (ct * 16 + l15) * 64 + (half ? 16 : 0);
      hb.q[0] = *(const uint4*)(base);
      hb.q[1] = *(const uint4*)(base + 32);
      acc[ct] = __builtin_amdgcn_wmma_f32_16x16x32_bf16(false, pa.v, false, hb.v,
                                                        (short)0, acc[ct], false, false);
    }
    __syncthreads();   // all waves done with hS[buf]/fS[buf] before re-staging
  }

  // ---- epilogue: out = gamma * (acc/l) + x at the flat-reinterpret index --
  const float gamma = gammap[0];
#pragma unroll
  for (int ct = 0; ct < 16; ++ct) {
#pragma unroll
    for (int r = 0; r < 8; ++r) {
      int i = i0 + r + 8 * half;
      int c = ct * 16 + l15;
      size_t idx = (size_t)b * NTOK * CH + (size_t)i * CH + c;
      out[idx] = gamma * (acc[ct][r] / l[r]) + x[idx];
    }
  }
}

// ---------------------------------------------------------------------------
extern "C" void kernel_launch(void* const* d_in, const int* in_sizes, int n_in,
                              void* d_out, int out_size, void* d_ws, size_t ws_size,
                              hipStream_t stream) {
  const float* x     = (const float*)d_in[0];
  const float* Wf    = (const float*)d_in[1];
  const float* bf_   = (const float*)d_in[2];
  const float* Wg    = (const float*)d_in[3];
  const float* bg_   = (const float*)d_in[4];
  const float* Wh    = (const float*)d_in[5];
  const float* bh_   = (const float*)d_in[6];
  const float* gamma = (const float*)d_in[7];
  float* out = (float*)d_out;

  char* ws = (char*)d_ws;
  // ws: y (B,N,64) bf16 | hJ (B,C,N) bf16 | Wcat (320,256) bf16 | bcat (320) f32
  size_t off_y  = 0;
  size_t off_h  = off_y + (size_t)NB * NTOK * YROW * 2;     //  2 MiB
  size_t off_w  = off_h + (size_t)NB * CH * NTOK * 2;       //  8 MiB
  size_t off_b  = off_w + (size_t)OTOT * CH * 2;            // 160 KiB
  bf16_t* yT   = (bf16_t*)(ws + off_y);
  bf16_t* hJ   = (bf16_t*)(ws + off_h);
  bf16_t* Wcat = (bf16_t*)(ws + off_w);
  float*  bcat = (float*) (ws + off_b);

  prep_weights<<<128, 256, 0, stream>>>(Wf, bf_, Wg, bg_, Wh, bh_, Wcat, bcat);
  proj_kernel <<<NB * (NTOK / 128), 256, 0, stream>>>(x, Wcat, bcat, yT, hJ);
  attn_kernel <<<NB * (NTOK / 128), 256, 0, stream>>>(x, yT, hJ, gamma, out);
}
